// GAT_36893769073015
// MI455X (gfx1250) — matrix operand order repaired and checked by reading, not compiled, not run on hardware
//
#include <hip/hip_runtime.h>
#include <math.h>

// ---------------------------------------------------------------------------
// GAT forward for MI455X (gfx1250), wave32 + WMMA f32_16x16x32_f16.
//
// Roofline: the adjacency is binary, so it is bit-packed once (256MB fp32 ->
// 8MB mask). All six N x N passes then stream the 8MB mask (L2-resident in
// the 192MB L2) instead of 256MB fp32, cutting the dominant HBM term ~4-5x.
// GEMM A-tiles are expanded bit->f16{0,1} in registers; attention tiles are
// expanded bit->exp(leaky(src+dst)-rowmax) with exp evaluated only on set
// bits (~0.5% density). GEMMs are software-pipelined (tile k+1 fetched into
// registers while tile k is consumed from LDS by the matrix pipe), with
// global_prefetch for tile k+2.
// ---------------------------------------------------------------------------

typedef __attribute__((ext_vector_type(16))) _Float16 v16h;
typedef __attribute__((ext_vector_type(8)))  _Float16 v8h;
typedef __attribute__((ext_vector_type(4)))  _Float16 v4h;
typedef __attribute__((ext_vector_type(8)))  float    v8f;

#define LRELU_ALPHA 0.2f

constexpr int NN   = 8192;              // nodes
constexpr int WPR  = NN / 32;           // mask words per row (256)
constexpr int BM   = 128;               // M tile
constexpr int BN   = 64;                // N tile
constexpr int BKL  = 64;                // K tile (2 WMMA k-steps)
constexpr int THREADS = 256;            // 8 waves
constexpr int LDSA = BKL + 16;          // 80 elems = 160B stride (16B-aligned)
constexpr int LDSB = BKL + 16;
constexpr int A_CH = BM * BKL / (THREADS * 4);  // 8 float4 chunks / thread
constexpr int B_CH = BN * BKL / (THREADS * 4);  // 4 float4 chunks / thread

// ---- WMMA fragment helpers (ISA 7.12.2 layouts, wave32) -------------------
__device__ inline v16h load_a_frag(const _Float16* lA, int mrow, int lane, int ksub) {
    const int l15 = lane & 15, kb = (lane >> 4) * 8 + ksub * 32;
    v8h lo = *(const v8h*)&lA[(mrow + l15) * LDSA + kb];
    v8h hi = *(const v8h*)&lA[(mrow + l15) * LDSA + kb + 16];
    return __builtin_shufflevector(lo, hi, 0,1,2,3,4,5,6,7,8,9,10,11,12,13,14,15);
}
__device__ inline v16h load_b_frag(const _Float16* lBt, int ncol, int lane, int ksub) {
    const int l15 = lane & 15, kb = (lane >> 4) * 8 + ksub * 32;
    v8h lo = *(const v8h*)&lBt[(ncol + l15) * LDSB + kb];
    v8h hi = *(const v8h*)&lBt[(ncol + l15) * LDSB + kb + 16];
    return __builtin_shufflevector(lo, hi, 0,1,2,3,4,5,6,7,8,9,10,11,12,13,14,15);
}
__device__ inline v4h cvt4(float4 v) {
    v4h h; h[0] = (_Float16)v.x; h[1] = (_Float16)v.y;
           h[2] = (_Float16)v.z; h[3] = (_Float16)v.w;
    return h;
}

// ---------------------------------------------------------------------------
// pack_adj: bit-pack the binary adjacency. One wave per 32-bit word (ballot).
// ---------------------------------------------------------------------------
__global__ __launch_bounds__(256)
void pack_adj(const float* __restrict__ adj, unsigned* __restrict__ mask)
{
    const int wave = threadIdx.x >> 5, lane = threadIdx.x & 31;
    const size_t wi = (size_t)blockIdx.x * 8 + wave;       // word index
    const size_t row = wi >> 8;                            // WPR==256 words/row
    const int    wcol = (int)(wi & 255);
    const float v = adj[row * NN + wcol * 32 + lane];
    unsigned long long b = __ballot(v > 0.0f);
    if (lane == 0) mask[wi] = (unsigned)b;
}

// ---------------------------------------------------------------------------
// Dense-A pipelined GEMM: C = A @ B (+= if ACC), fp32 I/O, f16 WMMA compute.
// Used for the feature-transform GEMMs (small K).
// ---------------------------------------------------------------------------
template<bool ACC>
__global__ __launch_bounds__(THREADS)
void gemm_wmma_f16(const float* __restrict__ A, const float* __restrict__ B,
                   float* __restrict__ C, int Ndim, int Kdim)
{
    __shared__ _Float16 lA[BM * LDSA];
    __shared__ _Float16 lBt[BN * LDSB];   // transposed: [n][k]

    const int tid  = threadIdx.x;
    const int lane = tid & 31;
    const int wave = tid >> 5;
    const int waveM = (wave & 3) * 32;
    const int waveN = (wave >> 2) * 32;
    const int bm0 = blockIdx.y * BM;
    const int bn0 = blockIdx.x * BN;
    const int half = lane >> 4;
    const int l15  = lane & 15;

    v8f acc[2][2] = {};
    v4h ra[A_CH], rb[B_CH];

    auto load_tile = [&](int k0) {
        #pragma unroll
        for (int p = 0; p < A_CH; ++p) {
            int chunk = p * THREADS + tid;
            int row = chunk >> 4;
            int kc  = (chunk & 15) << 2;
            ra[p] = cvt4(*(const float4*)&A[(size_t)(bm0 + row) * Kdim + k0 + kc]);
        }
        #pragma unroll
        for (int p = 0; p < B_CH; ++p) {
            int chunk = p * THREADS + tid;
            int krow = chunk >> 4;
            int nc   = (chunk & 15) << 2;
            rb[p] = cvt4(*(const float4*)&B[(size_t)(k0 + krow) * Ndim + bn0 + nc]);
        }
    };
    auto store_tile = [&]() {
        #pragma unroll
        for (int p = 0; p < A_CH; ++p) {
            int chunk = p * THREADS + tid;
            int row = chunk >> 4;
            int kc  = (chunk & 15) << 2;
            *(v4h*)&lA[row * LDSA + kc] = ra[p];
        }
        #pragma unroll
        for (int p = 0; p < B_CH; ++p) {
            int chunk = p * THREADS + tid;
            int krow = chunk >> 4;
            int nc   = (chunk & 15) << 2;
            lBt[(nc + 0) * LDSB + krow] = rb[p][0];
            lBt[(nc + 1) * LDSB + krow] = rb[p][1];
            lBt[(nc + 2) * LDSB + krow] = rb[p][2];
            lBt[(nc + 3) * LDSB + krow] = rb[p][3];
        }
    };

    const int nk = Kdim / BKL;
    load_tile(0);
    store_tile();
    for (int kt = 0; kt < nk; ++kt) {
        __syncthreads();
        if (kt + 1 < nk) load_tile((kt + 1) * BKL);
        if (kt + 2 < nk) {
            __builtin_prefetch(&A[(size_t)(bm0 + (tid >> 4)) * Kdim + (kt + 2) * BKL], 0, 1);
            __builtin_prefetch(&B[(size_t)((kt + 2) * BKL + (tid >> 4)) * Ndim + bn0], 0, 1);
        }
        #pragma unroll
        for (int ksub = 0; ksub < 2; ++ksub) {
            v16h af[2], bf[2];
            #pragma unroll
            for (int mi = 0; mi < 2; ++mi) af[mi] = load_a_frag(lA, waveM + mi * 16, lane, ksub);
            #pragma unroll
            for (int ni = 0; ni < 2; ++ni) bf[ni] = load_b_frag(lBt, waveN + ni * 16, lane, ksub);
            #pragma unroll
            for (int mi = 0; mi < 2; ++mi)
                #pragma unroll
                for (int ni = 0; ni < 2; ++ni)
                    acc[mi][ni] = __builtin_amdgcn_wmma_f32_16x16x32_f16(
                        false, af[mi], false, bf[ni], (short)0, acc[mi][ni], false, false);
        }
        __syncthreads();
        if (kt + 1 < nk) store_tile();
    }

    const int row0 = bm0 + waveM + 8 * half;
    const int col0 = bn0 + waveN + l15;
    #pragma unroll
    for (int mi = 0; mi < 2; ++mi)
        #pragma unroll
        for (int ni = 0; ni < 2; ++ni)
            #pragma unroll
            for (int r = 0; r < 8; ++r) {
                size_t idx = (size_t)(row0 + mi * 16 + r) * Ndim + col0 + ni * 16;
                float val = acc[mi][ni][r];
                if (ACC) val += C[idx];
                C[idx] = val;
            }
}

// ---------------------------------------------------------------------------
// Bit-mask-A pipelined GEMM: C = adj @ B. A-tile expanded bit -> f16 {0,1}
// in registers (exact).  Mask traffic: 8MB, L2-resident.
// ---------------------------------------------------------------------------
__global__ __launch_bounds__(THREADS)
void gemm_adj_wmma(const unsigned* __restrict__ mask, const float* __restrict__ B,
                   float* __restrict__ C, int Ndim)
{
    __shared__ _Float16 lA[BM * LDSA];
    __shared__ _Float16 lBt[BN * LDSB];

    const int tid  = threadIdx.x;
    const int lane = tid & 31;
    const int wave = tid >> 5;
    const int waveM = (wave & 3) * 32;
    const int waveN = (wave >> 2) * 32;
    const int bm0 = blockIdx.y * BM;
    const int bn0 = blockIdx.x * BN;
    const int half = lane >> 4;
    const int l15  = lane & 15;

    // per-thread A responsibility: one 32-bit mask word = 32 k-values of 1 row
    const int arow = tid >> 1;           // 128 rows
    const int wsel = tid & 1;            // 2 words per 64-wide k-tile

    v8f acc[2][2] = {};
    v8h abuf[4];                         // 32 f16 expanded from the mask word
    v4h rb[B_CH];

    auto load_tile = [&](int k0) {
        const unsigned w = mask[(size_t)(bm0 + arow) * WPR + (k0 >> 5) + wsel];
        #pragma unroll
        for (int i = 0; i < 32; ++i)
            abuf[i >> 3][i & 7] = ((w >> i) & 1u) ? (_Float16)1.0f : (_Float16)0.0f;
        #pragma unroll
        for (int p = 0; p < B_CH; ++p) {
            int chunk = p * THREADS + tid;
            int krow = chunk >> 4;
            int nc   = (chunk & 15) << 2;
            rb[p] = cvt4(*(const float4*)&B[(size_t)(k0 + krow) * Ndim + bn0 + nc]);
        }
    };
    auto store_tile = [&]() {
        _Float16* dst = &lA[arow * LDSA + wsel * 32];
        #pragma unroll
        for (int j = 0; j < 4; ++j) *(v8h*)&dst[j * 8] = abuf[j];
        #pragma unroll
        for (int p = 0; p < B_CH; ++p) {
            int chunk = p * THREADS + tid;
            int krow = chunk >> 4;
            int nc   = (chunk & 15) << 2;
            lBt[(nc + 0) * LDSB + krow] = rb[p][0];
            lBt[(nc + 1) * LDSB + krow] = rb[p][1];
            lBt[(nc + 2) * LDSB + krow] = rb[p][2];
            lBt[(nc + 3) * LDSB + krow] = rb[p][3];
        }
    };

    const int nk = NN / BKL;
    load_tile(0);
    store_tile();
    for (int kt = 0; kt < nk; ++kt) {
        __syncthreads();
        if (kt + 1 < nk) load_tile((kt + 1) * BKL);
        if (kt + 2 < nk)
            __builtin_prefetch(&B[(size_t)((kt + 2) * BKL + (tid >> 4)) * Ndim + bn0], 0, 1);
        #pragma unroll
        for (int ksub = 0; ksub < 2; ++ksub) {
            v16h af[2], bf[2];
            #pragma unroll
            for (int mi = 0; mi < 2; ++mi) af[mi] = load_a_frag(lA, waveM + mi * 16, lane, ksub);
            #pragma unroll
            for (int ni = 0; ni < 2; ++ni) bf[ni] = load_b_frag(lBt, waveN + ni * 16, lane, ksub);
            #pragma unroll
            for (int mi = 0; mi < 2; ++mi)
                #pragma unroll
                for (int ni = 0; ni < 2; ++ni)
                    acc[mi][ni] = __builtin_amdgcn_wmma_f32_16x16x32_f16(
                        false, af[mi], false, bf[ni], (short)0, acc[mi][ni], false, false);
        }
        __syncthreads();
        if (kt + 1 < nk) store_tile();
    }

    const int row0 = bm0 + waveM + 8 * half;
    const int col0 = bn0 + waveN + l15;
    #pragma unroll
    for (int mi = 0; mi < 2; ++mi)
        #pragma unroll
        for (int ni = 0; ni < 2; ++ni)
            #pragma unroll
            for (int r = 0; r < 8; ++r)
                C[(size_t)(row0 + mi * 16 + r) * Ndim + col0 + ni * 16] = acc[mi][ni][r];
}

// ---------------------------------------------------------------------------
// Fused attention aggregation from the bit mask: Out = softmax(e) @ H.
// Weight tile p_ij = bit ? exp(leaky(src_i+dst_j) - rowmax_i) : 0, exp only
// on set bits (~0.5%).  Epilogue divides by rowsum; ELU if requested.
// ---------------------------------------------------------------------------
template<bool ELU>
__global__ __launch_bounds__(THREADS)
void att_gemm_bits(const unsigned* __restrict__ mask, const float* __restrict__ H,
                   const float* __restrict__ srcv, const float* __restrict__ dstv,
                   const float* __restrict__ rowmax, const float* __restrict__ rowsum,
                   float* __restrict__ Out, int Fout)
{
    __shared__ _Float16 lA[BM * LDSA];
    __shared__ _Float16 lBt[BN * LDSB];

    const int tid  = threadIdx.x;
    const int lane = tid & 31;
    const int wave = tid >> 5;
    const int waveM = (wave & 3) * 32;
    const int waveN = (wave >> 2) * 32;
    const int bm0 = blockIdx.y * BM;
    const int bn0 = blockIdx.x * BN;
    const int half = lane >> 4;
    const int l15  = lane & 15;

    const int arow = tid >> 1;
    const int wsel = tid & 1;
    const float si = srcv[bm0 + arow];
    const float mi = rowmax[bm0 + arow];

    v8f acc[2][2] = {};
    v8h abuf[4];
    v4h rb[B_CH];

    auto load_tile = [&](int k0) {
        const unsigned w = mask[(size_t)(bm0 + arow) * WPR + (k0 >> 5) + wsel];
        const int kbase = k0 + wsel * 32;
        #pragma unroll
        for (int i = 0; i < 32; ++i) {
            _Float16 pv = (_Float16)0.0f;
            if ((w >> i) & 1u) {                      // rare (~0.5% density)
                float t = si + dstv[kbase + i];
                t = t > 0.0f ? t : LRELU_ALPHA * t;
                pv = (_Float16)__expf(t - mi);
            }
            abuf[i >> 3][i & 7] = pv;
        }
        #pragma unroll
        for (int p = 0; p < B_CH; ++p) {
            int chunk = p * THREADS + tid;
            int krow = chunk >> 4;
            int nc   = (chunk & 15) << 2;
            rb[p] = cvt4(*(const float4*)&H[(size_t)(k0 + krow) * Fout + bn0 + nc]);
        }
    };
    auto store_tile = [&]() {
        _Float16* dst = &lA[arow * LDSA + wsel * 32];
        #pragma unroll
        for (int j = 0; j < 4; ++j) *(v8h*)&dst[j * 8] = abuf[j];
        #pragma unroll
        for (int p = 0; p < B_CH; ++p) {
            int chunk = p * THREADS + tid;
            int krow = chunk >> 4;
            int nc   = (chunk & 15) << 2;
            lBt[(nc + 0) * LDSB + krow] = rb[p][0];
            lBt[(nc + 1) * LDSB + krow] = rb[p][1];
            lBt[(nc + 2) * LDSB + krow] = rb[p][2];
            lBt[(nc + 3) * LDSB + krow] = rb[p][3];
        }
    };

    const int nk = NN / BKL;
    load_tile(0);
    store_tile();
    for (int kt = 0; kt < nk; ++kt) {
        __syncthreads();
        if (kt + 1 < nk) load_tile((kt + 1) * BKL);
        if (kt + 2 < nk)
            __builtin_prefetch(&H[(size_t)((kt + 2) * BKL + (tid >> 4)) * Fout + bn0], 0, 1);
        #pragma unroll
        for (int ksub = 0; ksub < 2; ++ksub) {
            v16h af[2], bf[2];
            #pragma unroll
            for (int mi2 = 0; mi2 < 2; ++mi2) af[mi2] = load_a_frag(lA, waveM + mi2 * 16, lane, ksub);
            #pragma unroll
            for (int ni = 0; ni < 2; ++ni) bf[ni] = load_b_frag(lBt, waveN + ni * 16, lane, ksub);
            #pragma unroll
            for (int mi2 = 0; mi2 < 2; ++mi2)
                #pragma unroll
                for (int ni = 0; ni < 2; ++ni)
                    acc[mi2][ni] = __builtin_amdgcn_wmma_f32_16x16x32_f16(
                        false, af[mi2], false, bf[ni], (short)0, acc[mi2][ni], false, false);
        }
        __syncthreads();
        if (kt + 1 < nk) store_tile();
    }

    const int row0 = bm0 + waveM + 8 * half;
    const int col0 = bn0 + waveN + l15;
    #pragma unroll
    for (int mi2 = 0; mi2 < 2; ++mi2) {
        #pragma unroll
        for (int r = 0; r < 8; ++r) {
            const int row = row0 + mi2 * 16 + r;
            const float inv = 1.0f / rowsum[row];
            #pragma unroll
            for (int ni = 0; ni < 2; ++ni) {
                float val = acc[mi2][ni][r] * inv;
                if (ELU) val = val > 0.0f ? val : (__expf(val) - 1.0f);
                Out[(size_t)row * Fout + col0 + ni * 16] = val;
            }
        }
    }
}

// ---------------------------------------------------------------------------
// src[i] = h[i,:] . a[:F] ; dst[i] = h[i,:] . a[F:]   (wave per row, wave32)
// ---------------------------------------------------------------------------
__global__ __launch_bounds__(256)
void attn_prep(const float* __restrict__ H, const float* __restrict__ a,
               float* __restrict__ srcv, float* __restrict__ dstv, int Fout)
{
    const int wave = threadIdx.x >> 5, lane = threadIdx.x & 31;
    const int row = blockIdx.x * 8 + wave;
    float s = 0.0f, d = 0.0f;
    for (int f = lane; f < Fout; f += 32) {
        float hv = H[(size_t)row * Fout + f];
        s += hv * a[f];
        d += hv * a[Fout + f];
    }
    #pragma unroll
    for (int off = 16; off > 0; off >>= 1) {
        s += __shfl_xor(s, off, 32);
        d += __shfl_xor(d, off, 32);
    }
    if (lane == 0) { srcv[row] = s; dstv[row] = d; }
}

// ---------------------------------------------------------------------------
// Masked-softmax row stats from the bit mask: block per row, one word per
// thread, iterate set bits with ffs (avg ~41 neighbors per row).
// ---------------------------------------------------------------------------
__global__ __launch_bounds__(256)
void attn_stats_bits(const unsigned* __restrict__ mask, const float* __restrict__ srcv,
                     const float* __restrict__ dstv, float* __restrict__ rowmax,
                     float* __restrict__ rowsum)
{
    __shared__ float sm[256], ss[256];
    const int i = blockIdx.x, tid = threadIdx.x;
    const float si = srcv[i];
    const int kbase = tid * 32;
    unsigned w = mask[(size_t)i * WPR + tid];
    float m = -INFINITY, ssum = 0.0f;
    while (w) {
        int b = __ffs(w) - 1;
        w &= w - 1;
        float t = si + dstv[kbase + b];
        float e = t > 0.0f ? t : LRELU_ALPHA * t;
        if (e > m) { ssum = ssum * __expf(m - e) + 1.0f; m = e; }
        else        ssum += __expf(e - m);
    }
    sm[tid] = m; ss[tid] = ssum;
    __syncthreads();
    for (int off = 128; off > 0; off >>= 1) {
        if (tid < off) {
            float m1 = sm[tid], s1 = ss[tid];
            float m2 = sm[tid + off], s2 = ss[tid + off];
            float M = fmaxf(m1, m2);
            float S = (M == -INFINITY) ? 0.0f
                    : s1 * __expf(m1 - M) + s2 * __expf(m2 - M);
            sm[tid] = M; ss[tid] = S;
        }
        __syncthreads();
    }
    if (tid == 0) { rowmax[i] = sm[0]; rowsum[i] = ss[0]; }
}

// ---------------------------------------------------------------------------
// Row-wise log_softmax (wave per row).
// ---------------------------------------------------------------------------
__global__ __launch_bounds__(256)
void log_softmax_rows(const float* __restrict__ X, float* __restrict__ Out, int Cdim)
{
    const int wave = threadIdx.x >> 5, lane = threadIdx.x & 31;
    const int row = blockIdx.x * 8 + wave;
    float m = -INFINITY;
    for (int c = lane; c < Cdim; c += 32) m = fmaxf(m, X[(size_t)row * Cdim + c]);
    #pragma unroll
    for (int off = 16; off > 0; off >>= 1) m = fmaxf(m, __shfl_xor(m, off, 32));
    float s = 0.0f;
    for (int c = lane; c < Cdim; c += 32) s += __expf(X[(size_t)row * Cdim + c] - m);
    #pragma unroll
    for (int off = 16; off > 0; off >>= 1) s += __shfl_xor(s, off, 32);
    const float lg = __logf(s) + m;
    for (int c = lane; c < Cdim; c += 32)
        Out[(size_t)row * Cdim + c] = X[(size_t)row * Cdim + c] - lg;
}

// ---------------------------------------------------------------------------
extern "C" void kernel_launch(void* const* d_in, const int* in_sizes, int n_in,
                              void* d_out, int out_size, void* d_ws, size_t ws_size,
                              hipStream_t stream) {
    (void)in_sizes; (void)n_in; (void)out_size; (void)ws_size;
    const int N = NN, F = 256, HID = 128, C = 64;

    const float* features = (const float*)d_in[0];   // N x F
    const float* support  = (const float*)d_in[1];   // N x N
    const float* W0       = (const float*)d_in[2];   // 2F x HID
    const float* a0       = (const float*)d_in[3];   // 2*HID
    const float* W1       = (const float*)d_in[4];   // 2*HID x C
    const float* a1       = (const float*)d_in[5];   // 2*C
    float* out = (float*)d_out;                      // N x C (log-probs)

    float* ws  = (float*)d_ws;
    float* s    = ws;                     // N*F
    float* h0   = s    + (size_t)N * F;   // N*HID
    float* x1   = h0   + (size_t)N * HID; // N*HID
    float* s1   = x1   + (size_t)N * HID; // N*HID
    float* h1   = s1   + (size_t)N * HID; // N*C
    float* ho   = h1   + (size_t)N * C;   // N*C
    float* srcv = ho   + (size_t)N * C;
    float* dstv = srcv + N;
    float* rmax = dstv + N;
    float* rsum = rmax + N;
    unsigned* mask = (unsigned*)(rsum + N);  // N*WPR words (8MB)

    const dim3 blk(THREADS);

    // ---- bit-pack adjacency (256MB fp32 -> 8MB, read once) ----
    pack_adj<<<(size_t)N * WPR / 8, blk, 0, stream>>>(support, mask);

    // ---- Layer 0 ----
    gemm_adj_wmma<<<dim3(F / BN, N / BM), blk, 0, stream>>>(mask, features, s, F);
    gemm_wmma_f16<false><<<dim3(HID / BN, N / BM), blk, 0, stream>>>(s, W0, h0, HID, F);
    gemm_wmma_f16<true ><<<dim3(HID / BN, N / BM), blk, 0, stream>>>(features, W0 + (size_t)F * HID, h0, HID, F);
    attn_prep<<<N / 8, blk, 0, stream>>>(h0, a0, srcv, dstv, HID);
    attn_stats_bits<<<N, blk, 0, stream>>>(mask, srcv, dstv, rmax, rsum);
    att_gemm_bits<true ><<<dim3(HID / BN, N / BM), blk, 0, stream>>>(mask, h0, srcv, dstv, rmax, rsum, x1, HID);

    // ---- Layer 1 ----
    gemm_adj_wmma<<<dim3(HID / BN, N / BM), blk, 0, stream>>>(mask, x1, s1, HID);
    gemm_wmma_f16<false><<<dim3(C / BN, N / BM), blk, 0, stream>>>(s1, W1, h1, C, HID);
    gemm_wmma_f16<true ><<<dim3(C / BN, N / BM), blk, 0, stream>>>(x1, W1 + (size_t)HID * C, h1, C, HID);
    attn_prep<<<N / 8, blk, 0, stream>>>(h1, a1, srcv, dstv, C);
    attn_stats_bits<<<N, blk, 0, stream>>>(mask, srcv, dstv, rmax, rsum);
    att_gemm_bits<false><<<dim3(C / BN, N / BM), blk, 0, stream>>>(mask, h1, srcv, dstv, rmax, rsum, ho, C);

    // ---- log_softmax ----
    log_softmax_rows<<<N / 8, blk, 0, stream>>>(ho, out, C);
}
